// EncoderBlock_53206054863529
// MI455X (gfx1250) — matrix operand order repaired and checked by reading, compiled, not run on hardware
//
#include <hip/hip_runtime.h>
#include <math.h>
#include <stdint.h>

// ---------------------------------------------------------------------------
// EncoderBlock for MI455X (gfx1250): f16 WMMA GEMMs (TDM-staged, double
// buffered) + flash attention. All matrix math on v_wmma_f32_16x16x32_f16,
// fp32 accumulate. GEMM operand staging uses tensor_load_to_lds (TENSORcnt).
// ---------------------------------------------------------------------------

typedef __attribute__((ext_vector_type(16))) _Float16 v16h;
typedef __attribute__((ext_vector_type(8)))  _Float16 v8h;
typedef __attribute__((ext_vector_type(8)))  float    v8f;
typedef __attribute__((ext_vector_type(4)))  unsigned int u32x4;
typedef __attribute__((ext_vector_type(8)))  int      i32x8;
typedef __attribute__((ext_vector_type(4)))  int      i32x4;

#define LN_EPS 1e-5f

// ===========================================================================
// Tensor Data Mover: 2D tile load Global -> LDS.
//  - data_size = 2 bytes (f16)
//  - tile = tileCols x tileRows elements, row stride = rowStride elements
//  - LDS padding: rows are 16 DWORDs (32 halves); pad_interval=3 (16 DWORDs),
//    pad_amount enc: A uses 3 (4 DWORDs -> 40-half stride),
//                    B uses 7 (8 DWORDs -> 48-half stride)
// Descriptor layout per CDNA5 ISA ch.8 (D# groups 0/1; groups 2/3 zero = 2D).
// ===========================================================================
__device__ __forceinline__ void tdm_load_2d_f16(uint64_t gaddr, unsigned ldsAddr,
                                                unsigned rowStride,
                                                unsigned tileRows,
                                                unsigned padAmtEnc) {
  u32x4 g0;
  g0[0] = 1u;                                         // count=1, user mode
  g0[1] = ldsAddr;                                    // lds_addr (bytes)
  g0[2] = (unsigned)(gaddr & 0xffffffffu);            // global_addr[31:0]
  g0[3] = (unsigned)((gaddr >> 32) & 0x01ffffffu)     // global_addr[56:32]
        | (2u << 30);                                 // type = 2 ("image")
  i32x8 g1;
  g1[0] = (int)((1u << 16)                            // data_size = 2B
              | (1u << 20)                            // pad_enable
              | (3u << 22)                            // pad_interval: 16 DWORD
              | (padAmtEnc << 25));                   // pad_amount
  g1[1] = (int)((rowStride & 0xffffu) << 16);         // tensor_dim0[15:0]
  g1[2] = (int)(((rowStride >> 16) & 0xffffu)         // tensor_dim0[31:16]
              | ((tileRows & 0xffffu) << 16));        // tensor_dim1[15:0]
  g1[3] = (int)(32u << 16);                           // tile_dim0 = 32 elems
  g1[4] = (int)(tileRows & 0xffffu);                  // tile_dim1; tile_dim2=0
  g1[5] = (int)rowStride;                             // dim0_stride[31:0]
  g1[6] = 0;                                          // dim0_stride[47:32]
  g1[7] = 0;
  i32x4 z4 = {};
#if defined(__clang_major__) && (__clang_major__ >= 23)
  i32x8 z8 = {};
  __builtin_amdgcn_tensor_load_to_lds(g0, g1, z4, z4, z8, 0);
#else
  __builtin_amdgcn_tensor_load_to_lds(g0, g1, z4, z4, 0);
#endif
}

// ===========================================================================
// f32 -> f16 conversion (weights)
// ===========================================================================
__global__ void cvt_f32_to_f16(const float* __restrict__ in,
                               _Float16* __restrict__ out, size_t n) {
  size_t i = (size_t)blockIdx.x * blockDim.x + threadIdx.x;
  size_t stride = (size_t)gridDim.x * blockDim.x;
  for (; i < n; i += stride) out[i] = (_Float16)in[i];
}

// ===========================================================================
// LayerNorm (ddof=1, scalar alpha/bias, divide by (std+eps)) -> f16
// ===========================================================================
__global__ __launch_bounds__(256)
void layernorm_f16(const float* __restrict__ X, const float* __restrict__ alpha,
                   const float* __restrict__ beta, _Float16* __restrict__ Y,
                   int Dn) {
  const int row = blockIdx.x;
  const float* xr = X + (size_t)row * Dn;
  float s = 0.f, sq = 0.f;
  for (int i = threadIdx.x; i < Dn; i += blockDim.x) {
    float v = xr[i]; s += v; sq += v * v;
  }
  for (int off = 16; off > 0; off >>= 1) {
    s  += __shfl_xor(s,  off, 32);
    sq += __shfl_xor(sq, off, 32);
  }
  __shared__ float ss[8], ssq[8];
  int wid = threadIdx.x >> 5, lane = threadIdx.x & 31;
  if (lane == 0) { ss[wid] = s; ssq[wid] = sq; }
  __syncthreads();
  if (wid == 0) {
    s  = (lane < 8) ? ss[lane]  : 0.f;
    sq = (lane < 8) ? ssq[lane] : 0.f;
    for (int off = 4; off > 0; off >>= 1) {
      s  += __shfl_xor(s,  off, 32);
      sq += __shfl_xor(sq, off, 32);
    }
    if (lane == 0) { ss[0] = s; ssq[0] = sq; }
  }
  __syncthreads();
  s = ss[0]; sq = ssq[0];
  float mean = s / (float)Dn;
  float var  = (sq - (float)Dn * mean * mean) / (float)(Dn - 1);
  var = fmaxf(var, 0.f);
  float inv = alpha[0] / (sqrtf(var) + LN_EPS);
  float b   = beta[0];
  for (int i = threadIdx.x; i < Dn; i += blockDim.x)
    Y[(size_t)row * Dn + i] = (_Float16)((xr[i] - mean) * inv + b);
}

// ===========================================================================
// WMMA GEMM with TDM double-buffered staging:
//   C[M,N] = A[M,K] * Bw[N,K]^T + bias (+ resid)
// Block tile 128x64, 8 waves of 32x32, K-step 32. Wave 0 drives the TDM.
// ===========================================================================
#define BM 128
#define BN 64
#define BK 32
#define ASTR 40   // halves: 32 data + 8 pad (TDM pad_amount=3 -> 4 DWORDs)
#define BSTR 48   // halves: 32 data + 16 pad (TDM pad_amount=7 -> 8 DWORDs)

__global__ __launch_bounds__(256)
void gemm_f16_wmma(const _Float16* __restrict__ A, const _Float16* __restrict__ Bw,
                   const float* __restrict__ bias, const float* __restrict__ resid,
                   float* __restrict__ Cf, _Float16* __restrict__ Ch,
                   int M, int N, int K) {
  __shared__ __align__(16) _Float16 sA[2][BM * ASTR];
  __shared__ __align__(16) _Float16 sB[2][BN * BSTR];

  const int tid  = threadIdx.x;
  const int wid  = tid >> 5;
  const int lane = tid & 31;
  const int l16  = lane & 15;
  const int hi   = lane >> 4;

  const int bm0 = blockIdx.x * BM;
  const int bn0 = blockIdx.y * BN;
  const int wm0 = (wid & 3) * 32;
  const int wn0 = (wid >> 2) * 32;

  const uint64_t aAddr = (uint64_t)(uintptr_t)(A  + (size_t)bm0 * K);
  const uint64_t bAddr = (uint64_t)(uintptr_t)(Bw + (size_t)bn0 * K);
  const unsigned ldsA[2] = {(unsigned)(uintptr_t)&sA[0][0],
                            (unsigned)(uintptr_t)&sA[1][0]};
  const unsigned ldsB[2] = {(unsigned)(uintptr_t)&sB[0][0],
                            (unsigned)(uintptr_t)&sB[1][0]};

  v8f acc[2][2];
  v8f zero = {};
#pragma unroll
  for (int i = 0; i < 2; i++)
#pragma unroll
    for (int j = 0; j < 2; j++) acc[i][j] = zero;

  // prologue: stage first K tile into buffer 0
  if (wid == 0) {
    tdm_load_2d_f16(aAddr, ldsA[0], (unsigned)K, BM, 3u);
    tdm_load_2d_f16(bAddr, ldsB[0], (unsigned)K, BN, 7u);
  }

  const int nk = K / BK;
  for (int kt = 0; kt < nk; kt++) {
    const int cur = kt & 1;
    __builtin_amdgcn_s_wait_tensorcnt(0);   // wave0: tile arrived; others: nop
    __syncthreads();                        // publish LDS tile to all waves

    // issue next tile into the other buffer (freed by the barrier above)
    if (wid == 0 && kt + 1 < nk) {
      const uint64_t koff = (uint64_t)(kt + 1) * (BK * sizeof(_Float16));
      tdm_load_2d_f16(aAddr + koff, ldsA[cur ^ 1], (unsigned)K, BM, 3u);
      tdm_load_2d_f16(bAddr + koff, ldsB[cur ^ 1], (unsigned)K, BN, 7u);
    }

    // ---- fragments (ISA 16-bit A 16x32 / B 32x16 layouts) ----
    v16h af[2], bf[2];
#pragma unroll
    for (int mi = 0; mi < 2; mi++) {
      const int m  = wm0 + mi * 16 + l16;
      const int c0 = hi * 8;
      v8h lo = *(const v8h*)&sA[cur][m * ASTR + c0];
      v8h h8 = *(const v8h*)&sA[cur][m * ASTR + c0 + 16];
#pragma unroll
      for (int e = 0; e < 8; e++) { af[mi][e] = lo[e]; af[mi][e + 8] = h8[e]; }
    }
#pragma unroll
    for (int ni = 0; ni < 2; ni++) {
      const int n  = wn0 + ni * 16 + l16;
      const int c0 = hi * 8;
      v8h lo = *(const v8h*)&sB[cur][n * BSTR + c0];
      v8h h8 = *(const v8h*)&sB[cur][n * BSTR + c0 + 16];
#pragma unroll
      for (int e = 0; e < 8; e++) { bf[ni][e] = lo[e]; bf[ni][e + 8] = h8[e]; }
    }
#pragma unroll
    for (int mi = 0; mi < 2; mi++)
#pragma unroll
      for (int ni = 0; ni < 2; ni++)
        acc[mi][ni] = __builtin_amdgcn_wmma_f32_16x16x32_f16(
            false, af[mi], false, bf[ni], (short)0, acc[mi][ni], false, false);

    __syncthreads();  // all waves done reading `cur` before it is re-filled
  }

  // ---- epilogue: bias / residual / store ----
#pragma unroll
  for (int mi = 0; mi < 2; mi++) {
#pragma unroll
    for (int ni = 0; ni < 2; ni++) {
      const int col = bn0 + wn0 + ni * 16 + l16;
      const float bv = bias ? bias[col] : 0.f;
#pragma unroll
      for (int r = 0; r < 8; r++) {
        const int row = bm0 + wm0 + mi * 16 + r + hi * 8;
        const size_t idx = (size_t)row * N + col;
        float v = acc[mi][ni][r] + bv;
        if (resid) v += resid[idx];
        if (Cf) Cf[idx] = v;
        if (Ch) Ch[idx] = (_Float16)v;
      }
    }
  }
}

// ===========================================================================
// Flash attention (f16 in, f16 out, fp32 online softmax).
// grid = (S/64, H, B), 128 threads (4 waves); wave owns 16 q rows x DK=64.
// ===========================================================================
#define PSTR  72
#define VTSTR 72

__global__ __launch_bounds__(128)
void flash_attn(const _Float16* __restrict__ Q, const _Float16* __restrict__ Km,
                const _Float16* __restrict__ V, _Float16* __restrict__ O,
                int S, int Dmodel) {
  const int qt = blockIdx.x, h = blockIdx.y, b = blockIdx.z;
  const int tid = threadIdx.x, wid = tid >> 5, lane = tid & 31;
  const int l16 = lane & 15, hi = lane >> 4;

  const size_t base = ((size_t)b * S) * Dmodel + (size_t)h * 64;
  const _Float16* qp = Q + base;
  const _Float16* kp = Km + base;
  const _Float16* vp = V + base;
  _Float16* op = O + base;

  __shared__ __align__(16) _Float16 sVt[64 * VTSTR];
  __shared__ __align__(16) _Float16 sP[4][16 * PSTR];

  const int qr0 = qt * 64 + wid * 16;

  v16h aQ[2];
#pragma unroll
  for (int kc = 0; kc < 2; kc++) {
    const _Float16* qrow = qp + (size_t)(qr0 + l16) * Dmodel + kc * 32 + hi * 8;
    v8h lo = *(const v8h*)qrow;
    v8h h8 = *(const v8h*)(qrow + 16);
#pragma unroll
    for (int e = 0; e < 8; e++) { aQ[kc][e] = lo[e]; aQ[kc][e + 8] = h8[e]; }
  }

  v8f o[4];
  v8f zero = {};
  float mrow[8], lrow[8];
#pragma unroll
  for (int j = 0; j < 4; j++) o[j] = zero;
#pragma unroll
  for (int r = 0; r < 8; r++) { mrow[r] = -1e30f; lrow[r] = 0.f; }

  const int ntiles = S / 64;
  for (int t = 0; t < ntiles; t++) {
    const int kv0 = t * 64;
    __syncthreads();

    // stage V tile transposed into LDS: sVt[d][kv]
#pragma unroll
    for (int i = 0; i < 4; i++) {
      const int seg  = tid + 128 * i;
      const int vrow = seg >> 3;
      const int d0   = (seg & 7) * 8;
      v8h vv = *(const v8h*)(vp + (size_t)(kv0 + vrow) * Dmodel + d0);
#pragma unroll
      for (int e = 0; e < 8; e++) sVt[(d0 + e) * VTSTR + vrow] = vv[e];
    }

    // S = Q K^T / sqrt(DK); K fragments straight from global (contiguous)
    v8f sacc[4];
#pragma unroll
    for (int j = 0; j < 4; j++) {
      v8f c = zero;
#pragma unroll
      for (int kc = 0; kc < 2; kc++) {
        const _Float16* krow =
            kp + (size_t)(kv0 + j * 16 + l16) * Dmodel + kc * 32 + hi * 8;
        v8h lo = *(const v8h*)krow;
        v8h h8 = *(const v8h*)(krow + 16);
        v16h bf;
#pragma unroll
        for (int e = 0; e < 8; e++) { bf[e] = lo[e]; bf[e + 8] = h8[e]; }
        c = __builtin_amdgcn_wmma_f32_16x16x32_f16(false, aQ[kc], false, bf,
                                                   (short)0, c, false, false);
      }
      sacc[j] = c * 0.125f;
    }

    // online softmax (row reductions over 16-lane half-waves)
    float nm[8], fac[8];
#pragma unroll
    for (int r = 0; r < 8; r++) {
      float v0 = fmaxf(fmaxf(sacc[0][r], sacc[1][r]),
                       fmaxf(sacc[2][r], sacc[3][r]));
#pragma unroll
      for (int off = 1; off < 16; off <<= 1) v0 = fmaxf(v0, __shfl_xor(v0, off, 32));
      float m2 = fmaxf(mrow[r], v0);
      fac[r] = __expf(mrow[r] - m2);
      nm[r]  = m2;
    }
#pragma unroll
    for (int r = 0; r < 8; r++) {
      float rs = 0.f;
#pragma unroll
      for (int j = 0; j < 4; j++) {
        float p = __expf(sacc[j][r] - nm[r]);
        sacc[j][r] = p;
        rs += p;
      }
#pragma unroll
      for (int off = 1; off < 16; off <<= 1) rs += __shfl_xor(rs, off, 32);
      lrow[r] = lrow[r] * fac[r] + rs;
      mrow[r] = nm[r];
    }
#pragma unroll
    for (int j = 0; j < 4; j++)
#pragma unroll
      for (int r = 0; r < 8; r++) o[j][r] *= fac[r];

    // P (C-layout) -> LDS for re-read as A fragments
#pragma unroll
    for (int j = 0; j < 4; j++)
#pragma unroll
      for (int r = 0; r < 8; r++)
        sP[wid][(r + hi * 8) * PSTR + j * 16 + l16] = (_Float16)sacc[j][r];
    __syncthreads();

    // O += P @ V
    v16h aP[2];
#pragma unroll
    for (int kc = 0; kc < 2; kc++) {
      const _Float16* pr = &sP[wid][l16 * PSTR + kc * 32 + hi * 8];
      v8h lo = *(const v8h*)pr;
      v8h h8 = *(const v8h*)(pr + 16);
#pragma unroll
      for (int e = 0; e < 8; e++) { aP[kc][e] = lo[e]; aP[kc][e + 8] = h8[e]; }
    }
#pragma unroll
    for (int j = 0; j < 4; j++) {
#pragma unroll
      for (int kc = 0; kc < 2; kc++) {
        const _Float16* vr = &sVt[(j * 16 + l16) * VTSTR + kc * 32 + hi * 8];
        v8h lo = *(const v8h*)vr;
        v8h h8 = *(const v8h*)(vr + 16);
        v16h bf;
#pragma unroll
        for (int e = 0; e < 8; e++) { bf[e] = lo[e]; bf[e + 8] = h8[e]; }
        o[j] = __builtin_amdgcn_wmma_f32_16x16x32_f16(false, aP[kc], false, bf,
                                                      (short)0, o[j], false, false);
      }
    }
  }

#pragma unroll
  for (int j = 0; j < 4; j++) {
#pragma unroll
    for (int r = 0; r < 8; r++) {
      const int row = qr0 + r + hi * 8;
      op[(size_t)row * Dmodel + j * 16 + l16] = (_Float16)(o[j][r] / lrow[r]);
    }
  }
}

// ===========================================================================
// Launcher
// ===========================================================================
extern "C" void kernel_launch(void* const* d_in, const int* in_sizes, int n_in,
                              void* d_out, int out_size, void* d_ws, size_t ws_size,
                              hipStream_t stream) {
  (void)in_sizes; (void)n_in; (void)out_size; (void)ws_size;
  const int Bb = 2, Ss = 2048, Dd = 1024, Hh = 16, Ff = 4096;
  const int Mr = Bb * Ss;

  const float* x      = (const float*)d_in[0];
  const float* wq     = (const float*)d_in[2];
  const float* bq     = (const float*)d_in[3];
  const float* wk     = (const float*)d_in[4];
  const float* bk     = (const float*)d_in[5];
  const float* wv     = (const float*)d_in[6];
  const float* bv     = (const float*)d_in[7];
  const float* wo     = (const float*)d_in[8];
  const float* bo     = (const float*)d_in[9];
  const float* w1     = (const float*)d_in[10];
  const float* b1     = (const float*)d_in[11];
  const float* w2     = (const float*)d_in[12];
  const float* b2     = (const float*)d_in[13];
  const float* alpha1 = (const float*)d_in[14];
  const float* bias1  = (const float*)d_in[15];
  const float* alpha2 = (const float*)d_in[16];
  const float* bias2  = (const float*)d_in[17];
  float* out = (float*)d_out;

  char* ws = (char*)d_ws;
  size_t off = 0;
  auto alloc = [&](size_t bytes) -> char* {
    char* p = ws + off;
    off += (bytes + 255) & ~(size_t)255;
    return p;
  };
  _Float16* wqh  = (_Float16*)alloc((size_t)Dd * Dd * 2);
  _Float16* wkh  = (_Float16*)alloc((size_t)Dd * Dd * 2);
  _Float16* wvh  = (_Float16*)alloc((size_t)Dd * Dd * 2);
  _Float16* woh  = (_Float16*)alloc((size_t)Dd * Dd * 2);
  _Float16* w1h  = (_Float16*)alloc((size_t)Ff * Dd * 2);
  _Float16* w2h  = (_Float16*)alloc((size_t)Dd * Ff * 2);
  float*    h1   = (float*)   alloc((size_t)Mr * Dd * 4);
  _Float16* n1h  = (_Float16*)alloc((size_t)Mr * Dd * 2);
  _Float16* qh   = (_Float16*)alloc((size_t)Mr * Dd * 2);
  _Float16* kh   = (_Float16*)alloc((size_t)Mr * Dd * 2);
  _Float16* vh   = (_Float16*)alloc((size_t)Mr * Dd * 2);
  _Float16* midh = (_Float16*)alloc((size_t)Mr * Ff * 2);
  _Float16* attnh = n1h;  // dead after QKV gemms
  _Float16* n2h   = qh;   // dead after flash attention

  const size_t nDD = (size_t)Dd * Dd, nFD = (size_t)Ff * Dd;
  cvt_f32_to_f16<<<2048, 256, 0, stream>>>(wq, wqh, nDD);
  cvt_f32_to_f16<<<2048, 256, 0, stream>>>(wk, wkh, nDD);
  cvt_f32_to_f16<<<2048, 256, 0, stream>>>(wv, wvh, nDD);
  cvt_f32_to_f16<<<2048, 256, 0, stream>>>(wo, woh, nDD);
  cvt_f32_to_f16<<<2048, 256, 0, stream>>>(w1, w1h, nFD);
  cvt_f32_to_f16<<<2048, 256, 0, stream>>>(w2, w2h, nFD);

  layernorm_f16<<<Mr, 256, 0, stream>>>(x, alpha1, bias1, n1h, Dd);

  dim3 gP(Mr / BM, Dd / BN);
  gemm_f16_wmma<<<gP, 256, 0, stream>>>(n1h, wqh, bq, nullptr, nullptr, qh, Mr, Dd, Dd);
  gemm_f16_wmma<<<gP, 256, 0, stream>>>(n1h, wkh, bk, nullptr, nullptr, kh, Mr, Dd, Dd);
  gemm_f16_wmma<<<gP, 256, 0, stream>>>(n1h, wvh, bv, nullptr, nullptr, vh, Mr, Dd, Dd);

  dim3 gF(Ss / 64, Hh, Bb);
  flash_attn<<<gF, 128, 0, stream>>>(qh, kh, vh, attnh, Ss, Dd);

  gemm_f16_wmma<<<gP, 256, 0, stream>>>(attnh, woh, bo, x, h1, nullptr, Mr, Dd, Dd);

  layernorm_f16<<<Mr, 256, 0, stream>>>(h1, alpha2, bias2, n2h, Dd);

  dim3 g1(Mr / BM, Ff / BN);
  gemm_f16_wmma<<<g1, 256, 0, stream>>>(n2h, w1h, b1, nullptr, nullptr, midh, Mr, Ff, Dd);
  gemm_f16_wmma<<<gP, 256, 0, stream>>>(midh, w2h, b2, h1, out, nullptr, Mr, Dd, Ff);
}
